// MultiHeadAttention_87127706566924
// MI455X (gfx1250) — matrix operand order repaired and checked by reading, not compiled, not run on hardware
//
#include <hip/hip_runtime.h>
#include <hip/hip_bf16.h>

// Problem constants (match reference).
#define D_DIM  2048
#define B_ROWS 32768
#define HEADS  16
#define HDV    128
#define LN_EPS 1e-5f
#define KSTEP  32
#define NKIT   (D_DIM / KSTEP)   // 64

typedef __attribute__((ext_vector_type(16))) __bf16 v16bf;
typedef __attribute__((ext_vector_type(2)))  __bf16 v2bf;
typedef __attribute__((ext_vector_type(8)))  float  v8f;

union Frag16 {
    v16bf bf;
    uint4 q[2];
};

// Native f32->bf16 (RNE) via hardware convert; compiler packs pairs.
__device__ __forceinline__ unsigned short f2bf(float f) {
    union { __bf16 b; unsigned short u; } c;
    c.b = (__bf16)f;
    return c.u;
}
__device__ __forceinline__ unsigned int pack2(float a, float b) {
    union { v2bf v; unsigned int u; } c;
    c.v[0] = (__bf16)a;
    c.v[1] = (__bf16)b;
    return c.u;
}

// ---------------------------------------------------------------------------
// Kernel 1: W [k][n] f32 row-major  ->  Wt [n][k] bf16 row-major
// ---------------------------------------------------------------------------
__global__ __launch_bounds__(256) void wtrans_kernel(const float* __restrict__ W,
                                                     unsigned short* __restrict__ Wt) {
    __shared__ float tile[32][33];
    const int tx = threadIdx.x, ty = threadIdx.y;
    const int bn = blockIdx.x * 32;  // n (output) dim
    const int bk = blockIdx.y * 32;  // k (input) dim
#pragma unroll
    for (int j = 0; j < 32; j += 8)
        tile[ty + j][tx] = W[(size_t)(bk + ty + j) * D_DIM + bn + tx];
    __syncthreads();
#pragma unroll
    for (int j = 0; j < 32; j += 8)
        Wt[(size_t)(bn + ty + j) * D_DIM + bk + tx] = f2bf(tile[tx][ty + j]);
}

// ---------------------------------------------------------------------------
// Kernel 2: C[M,N] = A[M,K](f32) @ Wt[N,K](bf16)^T + bias, f32 accumulate.
// Block tile 128(M) x 256(N); 8 waves as 2(M) x 4(N); wave tile 64x64
// (4x4 WMMA 16x16 tiles). K-step 32, double-buffered LDS (ping-pong),
// one barrier per K-step; next tile's global loads overlap the 16 WMMAs.
// LDS row stride 48 elems (96 B) keeps ds_read_b128 16B-aligned.
// ---------------------------------------------------------------------------
__global__ __launch_bounds__(256) void gemm_bf16_kernel(const float* __restrict__ A,
                                                        const unsigned short* __restrict__ Bt,
                                                        const float* __restrict__ bias,
                                                        float* __restrict__ C) {
    __shared__ unsigned short As[2][128 * 48];   // 12 KB x2
    __shared__ unsigned short Bs[2][256 * 48];   // 24 KB x2

    const int tid  = threadIdx.x;
    const int lane = tid & 31;
    const int wid  = tid >> 5;
    const int wm   = wid & 1;    // 0..1 -> M offset wm*64
    const int wn   = wid >> 1;   // 0..3 -> N offset wn*64
    const int bm   = blockIdx.y * 128;
    const int bn   = blockIdx.x * 256;

    // per-thread global->LDS staging coordinates
    const int ar = tid >> 3, ac = (tid & 7) * 4;  // A: 4 rows (ar+32i), 4 f32 each
    const int br = tid >> 2, bc = (tid & 3) * 8;  // B: 4 rows (br+64i), 8 bf16 each

    v8f acc[4][4];
#pragma unroll
    for (int mi = 0; mi < 4; ++mi)
#pragma unroll
        for (int ni = 0; ni < 4; ++ni)
            acc[mi][ni] = {};

    float4 ga[4];
    uint4  gb[4];

    // ---- prologue: fetch + stage tile 0 into buffer 0 ----
#pragma unroll
    for (int i = 0; i < 4; ++i)
        ga[i] = *(const float4*)&A[(size_t)(bm + ar + 32 * i) * D_DIM + ac];
#pragma unroll
    for (int i = 0; i < 4; ++i)
        gb[i] = *(const uint4*)&Bt[(size_t)(bn + br + 64 * i) * D_DIM + bc];
#pragma unroll
    for (int i = 0; i < 4; ++i) {
        uint2 p; p.x = pack2(ga[i].x, ga[i].y); p.y = pack2(ga[i].z, ga[i].w);
        *(uint2*)&As[0][(ar + 32 * i) * 48 + ac] = p;
    }
#pragma unroll
    for (int i = 0; i < 4; ++i)
        *(uint4*)&Bs[0][(br + 64 * i) * 48 + bc] = gb[i];
    __syncthreads();

    for (int kt = 0; kt < NKIT; ++kt) {
        const int buf = kt & 1;

        // ---- issue next tile's global loads early (overlap with WMMA) ----
        if (kt + 1 < NKIT) {
            const int k0 = (kt + 1) * KSTEP;
#pragma unroll
            for (int i = 0; i < 4; ++i)
                ga[i] = *(const float4*)&A[(size_t)(bm + ar + 32 * i) * D_DIM + k0 + ac];
#pragma unroll
            for (int i = 0; i < 4; ++i)
                gb[i] = *(const uint4*)&Bt[(size_t)(bn + br + 64 * i) * D_DIM + k0 + bc];
        }

        // ---- fragment loads per ISA 16-bit A(16x32)/B(32x16) layouts ----
        Frag16 a[4], b[4];
        const int kA = (lane >> 4) * 8;   // lanes 0-15: K 0..7/16..23; 16-31: K 8..15/24..31
        const int kB = (lane >> 4) * 8;   // lanes 0-15: K 0..7/16..23; 16-31: K 8..15/24..31
#pragma unroll
        for (int mi = 0; mi < 4; ++mi) {
            const int r = wm * 64 + mi * 16 + (lane & 15);
            a[mi].q[0] = *(const uint4*)&As[buf][r * 48 + kA];
            a[mi].q[1] = *(const uint4*)&As[buf][r * 48 + kA + 16];
        }
#pragma unroll
        for (int ni = 0; ni < 4; ++ni) {
            const int r = wn * 64 + ni * 16 + (lane & 15);
            b[ni].q[0] = *(const uint4*)&Bs[buf][r * 48 + kB];
            b[ni].q[1] = *(const uint4*)&Bs[buf][r * 48 + kB + 16];
        }

#pragma unroll
        for (int mi = 0; mi < 4; ++mi)
#pragma unroll
            for (int ni = 0; ni < 4; ++ni)
                acc[mi][ni] = __builtin_amdgcn_wmma_f32_16x16x32_bf16(
                    false, a[mi].bf, false, b[ni].bf, (short)0, acc[mi][ni], false, false);

        // ---- stage next tile into the other buffer; single barrier ----
        if (kt + 1 < NKIT) {
            const int nbuf = buf ^ 1;
#pragma unroll
            for (int i = 0; i < 4; ++i) {
                uint2 p; p.x = pack2(ga[i].x, ga[i].y); p.y = pack2(ga[i].z, ga[i].w);
                *(uint2*)&As[nbuf][(ar + 32 * i) * 48 + ac] = p;
            }
#pragma unroll
            for (int i = 0; i < 4; ++i)
                *(uint4*)&Bs[nbuf][(br + 64 * i) * 48 + bc] = gb[i];
            __syncthreads();
        }
    }

    // Epilogue: C/D layout: VGPR r holds (M = r + 8*(lane/16), N = lane%16)
#pragma unroll
    for (int ni = 0; ni < 4; ++ni) {
        const int col = bn + wn * 64 + ni * 16 + (lane & 15);
        const float bvv = bias[col];
#pragma unroll
        for (int mi = 0; mi < 4; ++mi) {
#pragma unroll
            for (int r = 0; r < 8; ++r) {
                const int row = bm + wm * 64 + mi * 16 + r + 8 * (lane >> 4);
                C[(size_t)row * D_DIM + col] = acc[mi][ni][r] + bvv;
            }
        }
    }
}

// ---------------------------------------------------------------------------
// Kernel 3: per-sample cosine attention across heads (no softmax).
// One block per sample. Writes head-transposed output: attn[b, hd*H + h].
// attn may alias kp (row-local: kp row fully consumed into LDS first).
// ---------------------------------------------------------------------------
__global__ __launch_bounds__(256) void attn_kernel(const float* __restrict__ qp,
                                                   const float* __restrict__ kp,
                                                   const float* __restrict__ vp,
                                                   float* __restrict__ attn) {
    __shared__ float qs[HEADS * HDV], ks[HEADS * HDV], vs[HEADS * HDV];
    __shared__ float ss[HEADS * HEADS];
    __shared__ float rq[HEADS], rk[HEADS];
    const int t = threadIdx.x;
    const size_t base = (size_t)blockIdx.x * D_DIM;

#pragma unroll
    for (int i = 0; i < 2; ++i) {
        const int e = (t + 256 * i) * 4;
        *(float4*)&qs[e] = *(const float4*)&qp[base + e];
        *(float4*)&ks[e] = *(const float4*)&kp[base + e];
        *(float4*)&vs[e] = *(const float4*)&vp[base + e];
    }
    __syncthreads();

    if (t < 32) {
        const float* src = (t < 16) ? qs : ks;
        const int h = t & 15;
        float s = 0.f;
        for (int d = 0; d < HDV; ++d) { const float x = src[h * HDV + d]; s += x * x; }
        const float r = rsqrtf(s);
        if (t < 16) rq[h] = r; else rk[h] = r;
    }
    __syncthreads();

    {   // scores: thread t -> (h = t/16, g = t%16)
        const int h = t >> 4, g = t & 15;
        float s = 0.f;
        for (int d = 0; d < HDV; ++d) s += qs[h * HDV + d] * ks[g * HDV + d];
        ss[h * 16 + g] = s * rq[h] * rk[g] * (1.0f / HDV);
    }
    __syncthreads();

    {   // o[h][d] = sum_g s[h][g] * v[g][d]; write transposed (d*H + h)
        const int h = t >> 4, d0 = (t & 15) * 8;
        float acc[8] = {0.f, 0.f, 0.f, 0.f, 0.f, 0.f, 0.f, 0.f};
        for (int g = 0; g < 16; ++g) {
            const float sv = ss[h * 16 + g];
#pragma unroll
            for (int j = 0; j < 8; ++j) acc[j] += sv * vs[g * HDV + d0 + j];
        }
#pragma unroll
        for (int j = 0; j < 8; ++j)
            attn[base + (size_t)(d0 + j) * HEADS + h] = acc[j];
    }
}

// ---------------------------------------------------------------------------
// Kernel 4: out = LayerNorm(qp + o2) * gamma + beta  (in-place safe on o2/out)
// ---------------------------------------------------------------------------
__global__ __launch_bounds__(256) void ln_kernel(const float* __restrict__ qp,
                                                 const float* __restrict__ o2,
                                                 const float* __restrict__ gamma,
                                                 const float* __restrict__ beta,
                                                 float* __restrict__ out) {
    __shared__ float xs[D_DIM];
    __shared__ float r1[256], r2[256];
    const int t = threadIdx.x;
    const size_t base = (size_t)blockIdx.x * D_DIM;
    float s = 0.f, s2 = 0.f;
#pragma unroll
    for (int i = 0; i < 8; ++i) {
        const int idx = t + 256 * i;
        const float x = qp[base + idx] + o2[base + idx];
        xs[idx] = x; s += x; s2 += x * x;
    }
    r1[t] = s; r2[t] = s2;
    __syncthreads();
    for (int off = 128; off > 0; off >>= 1) {
        if (t < off) { r1[t] += r1[t + off]; r2[t] += r2[t + off]; }
        __syncthreads();
    }
    const float mu   = r1[0] * (1.0f / D_DIM);
    const float var  = r2[0] * (1.0f / D_DIM) - mu * mu;
    const float rstd = rsqrtf(var + LN_EPS);
#pragma unroll
    for (int i = 0; i < 8; ++i) {
        const int idx = t + 256 * i;
        out[base + idx] = (xs[idx] - mu) * rstd * gamma[idx] + beta[idx];
    }
}

// ---------------------------------------------------------------------------
extern "C" void kernel_launch(void* const* d_in, const int* in_sizes, int n_in,
                              void* d_out, int out_size, void* d_ws, size_t ws_size,
                              hipStream_t stream) {
    const float* q     = (const float*)d_in[0];
    const float* k     = (const float*)d_in[1];
    const float* v     = (const float*)d_in[2];
    const float* Wq    = (const float*)d_in[3];
    const float* bq    = (const float*)d_in[4];
    const float* Wk    = (const float*)d_in[5];
    const float* bk    = (const float*)d_in[6];
    const float* Wv    = (const float*)d_in[7];
    const float* bv    = (const float*)d_in[8];
    const float* Wo    = (const float*)d_in[9];
    const float* bo    = (const float*)d_in[10];
    const float* gamma = (const float*)d_in[11];
    const float* beta  = (const float*)d_in[12];
    float* out = (float*)d_out;

    // Workspace layout (needs 4*8MB + 3*256MB = 800MB):
    char* ws = (char*)d_ws;
    const size_t WT  = (size_t)D_DIM * D_DIM * sizeof(unsigned short); // 8 MB
    const size_t MAT = (size_t)B_ROWS * D_DIM * sizeof(float);         // 256 MB
    unsigned short* wtq = (unsigned short*)(ws);
    unsigned short* wtk = (unsigned short*)(ws + WT);
    unsigned short* wtv = (unsigned short*)(ws + 2 * WT);
    unsigned short* wto = (unsigned short*)(ws + 3 * WT);
    float* qp = (float*)(ws + 4 * WT);
    float* kp = (float*)(ws + 4 * WT + MAT);
    float* vp = (float*)(ws + 4 * WT + 2 * MAT);

    // 1) transpose+convert weights to bf16
    dim3 tgrid(D_DIM / 32, D_DIM / 32), tblk(32, 8);
    wtrans_kernel<<<tgrid, tblk, 0, stream>>>(Wq, wtq);
    wtrans_kernel<<<tgrid, tblk, 0, stream>>>(Wk, wtk);
    wtrans_kernel<<<tgrid, tblk, 0, stream>>>(Wv, wtv);
    wtrans_kernel<<<tgrid, tblk, 0, stream>>>(Wo, wto);

    // 2) q/k/v projections (WMMA bf16, f32 accumulate)
    dim3 ggrid(D_DIM / 256, B_ROWS / 128);
    gemm_bf16_kernel<<<ggrid, 256, 0, stream>>>(q, wtq, bq, qp);
    gemm_bf16_kernel<<<ggrid, 256, 0, stream>>>(k, wtk, bk, kp);
    gemm_bf16_kernel<<<ggrid, 256, 0, stream>>>(v, wtv, bv, vp);

    // 3) per-sample head attention; result (head-transposed) overwrites kp
    attn_kernel<<<B_ROWS, 256, 0, stream>>>(qp, kp, vp, kp);

    // 4) output projection into d_out
    gemm_bf16_kernel<<<ggrid, 256, 0, stream>>>(kp, wto, bo, out);

    // 5) residual + layernorm (in-place on d_out)
    ln_kernel<<<B_ROWS, 256, 0, stream>>>(qp, out, gamma, beta, out);
}
